// TernaryLinear_87608742904344
// MI455X (gfx1250) — matrix operand order repaired and checked by reading, not compiled, not run on hardware
//
#include <hip/hip_runtime.h>
#include <stdint.h>

typedef __attribute__((ext_vector_type(16))) __bf16 v16bf;
typedef __attribute__((ext_vector_type(8)))  float  v8f;
typedef int v4i __attribute__((__vector_size__(16)));

#define M_DIM    1024
#define N_DIM    4096
#define K_DIM    4096
#define NBLK     32

#define BM       128
#define BN       128
#define BKT      2                 /* k-tiles (of K=32) per LDS stage -> K chunk 64 */
#define FRAGB    1024              /* bytes per 16x32 (or 32x16) bf16 fragment     */
#define KT_TOT   (K_DIM / 32)      /* 128 k-tiles                                  */
#define NSTAGES  (KT_TOT / BKT)    /* 64 stages                                    */
#define STAGEB   (8 * BKT * FRAGB) /* 16 KB per stage for A, same for B            */

#if defined(__AMDGCN__) && __has_builtin(__builtin_amdgcn_global_load_async_to_lds_b128) && __has_builtin(__builtin_amdgcn_s_wait_asynccnt)
#define USE_ASYNC 1
#else
#define USE_ASYNC 0
#endif

#define GLOBAL_AS __attribute__((address_space(1)))
#define LDS_AS    __attribute__((address_space(3)))

// ---------------------------------------------------------------------------
// 16-byte global -> LDS copy. Async (ASYNCcnt path) if available.
// ---------------------------------------------------------------------------
__device__ __forceinline__ void copy16(const uint8_t* g, uint8_t* l) {
#if USE_ASYNC
  __builtin_amdgcn_global_load_async_to_lds_b128(
      (GLOBAL_AS v4i*)(uintptr_t)g,
      (LDS_AS v4i*)(uintptr_t)(uint32_t)(uintptr_t)l,  /* low 32b = LDS addr */
      0, 0);
#else
  *(uint4*)l = *(const uint4*)g;
#endif
}

// ---------------------------------------------------------------------------
// prep_x: q = perm(perm(k)); Xg[m,k] = bf16(x[m, q[k]]) in A-fragment order.
// A 16x32 bf16 layout (ISA 7.12.2): lane L holds row M=L%16; element e maps to
//   k_local = (e<8 ? e : e+8) + (L>=16 ? 8 : 0)
// Fragment (m_tile, k_tile) is 32 lanes x 16 bf16 = 1KB contiguous.
// ---------------------------------------------------------------------------
__global__ __launch_bounds__(256) void prep_x_kernel(
    const float* __restrict__ x, const long long* __restrict__ perm,
    __bf16* __restrict__ XgF) {
  int t     = blockIdx.x * 256 + threadIdx.x;   // 64*128*32 threads
  int lane  = t & 31;
  int ktile = (t >> 5) & (KT_TOT - 1);
  int mtile = t >> 12;
  int m     = mtile * 16 + (lane & 15);
  int khi   = (lane >= 16) ? 8 : 0;

  __align__(32) __bf16 vals[16];
#pragma unroll
  for (int e = 0; e < 16; ++e) {
    int klocal = ((e < 8) ? e : e + 8) + khi;
    int k      = ktile * 32 + klocal;
    int q      = (int)perm[(int)perm[k]];
    vals[e]    = (__bf16)x[(size_t)m * K_DIM + q];
  }
  __bf16* dst = XgF + ((size_t)(mtile * KT_TOT + ktile) * 32 + lane) * 16;
  *(v16bf*)dst = *(const v16bf*)vals;
}

// ---------------------------------------------------------------------------
// prep_w: Wb[o,k] = bf16(alpha[o,k/128]*T[o,k] + mu[o,k/128]) in B-fragment
// order. B 32x16 bf16 layout: lane L holds col N=L%16; element e maps to
//   k_local = (e<8 ? e : e+8) + (L>=16 ? 8 : 0)
// ---------------------------------------------------------------------------
__global__ __launch_bounds__(256) void prep_w_kernel(
    const int* __restrict__ T, const float* __restrict__ alpha,
    const float* __restrict__ mu, __bf16* __restrict__ WbF) {
  int t     = blockIdx.x * 256 + threadIdx.x;   // 256*128*32 threads
  int lane  = t & 31;
  int ktile = (t >> 5) & (KT_TOT - 1);
  int ntile = t >> 12;
  int o     = ntile * 16 + (lane & 15);
  int kbase = ktile * 32 + ((lane >= 16) ? 8 : 0);

  __align__(32) __bf16 vals[16];
#pragma unroll
  for (int e = 0; e < 16; ++e) {
    int k    = kbase + ((e < 8) ? e : e + 8);
    int blk  = k >> 7;
    float a  = alpha[o * NBLK + blk];
    float mm = mu[o * NBLK + blk];
    vals[e]  = (__bf16)(a * (float)T[(size_t)o * K_DIM + k] + mm);
  }
  __bf16* dst = WbF + ((size_t)(ntile * KT_TOT + ktile) * 32 + lane) * 16;
  *(v16bf*)dst = *(const v16bf*)vals;
}

// ---------------------------------------------------------------------------
// GEMM compute for one LDS stage: 2 k-steps x (4 M-frag x 2 N-frag) WMMAs.
// ---------------------------------------------------------------------------
__device__ __forceinline__ void compute_stage(const uint8_t* bufA,
                                              const uint8_t* bufB, int wm,
                                              int wn, int lane, v8f c[4][2]) {
#pragma unroll
  for (int kt = 0; kt < BKT; ++kt) {
    v16bf a[4], b[2];
#pragma unroll
    for (int i = 0; i < 4; ++i) {
      int mf = wm * 4 + i;
      a[i] = *(const v16bf*)(bufA + (mf * BKT + kt) * FRAGB + lane * 32);
    }
#pragma unroll
    for (int j = 0; j < 2; ++j) {
      int nf = wn * 2 + j;
      b[j] = *(const v16bf*)(bufB + (nf * BKT + kt) * FRAGB + lane * 32);
    }
#pragma unroll
    for (int i = 0; i < 4; ++i)
#pragma unroll
      for (int j = 0; j < 2; ++j)
        c[i][j] = __builtin_amdgcn_wmma_f32_16x16x32_bf16(
            false, a[i], false, b[j], (short)0, c[i][j], false, false);
  }
}

// ---------------------------------------------------------------------------
// Main GEMM: 128x128 block, 256 threads (8 waves as 2x4), double-buffered LDS.
// ---------------------------------------------------------------------------
__global__ __launch_bounds__(256) void gemm_kernel(
    const __bf16* __restrict__ XgF, const __bf16* __restrict__ WbF,
    const float* __restrict__ bias, float* __restrict__ outp) {
#if USE_ASYNC
  __shared__ __align__(32) uint8_t lds[2 * 2 * STAGEB];  // 64 KB (2 bufs x A+B)
#else
  __shared__ __align__(32) uint8_t lds[2 * STAGEB];      // 32 KB (A+B)
#endif
  const int tid     = threadIdx.x;
  const int lane    = tid & 31;
  const int wave    = tid >> 5;
  const int wm      = wave >> 2;  // 0..1 -> 64 rows
  const int wn      = wave & 3;   // 0..3 -> 32 cols
  const int m_block = blockIdx.y;
  const int n_block = blockIdx.x;

  const uint8_t* gA = (const uint8_t*)XgF;
  const uint8_t* gB = (const uint8_t*)WbF;

  // Accumulators initialized with bias: lane%16 is the output column.
  v8f c[4][2];
  {
    float bv0 = bias[n_block * BN + (wn * 2 + 0) * 16 + (lane & 15)];
    float bv1 = bias[n_block * BN + (wn * 2 + 1) * 16 + (lane & 15)];
#pragma unroll
    for (int i = 0; i < 4; ++i)
#pragma unroll
      for (int r = 0; r < 8; ++r) {
        c[i][0][r] = bv0;
        c[i][1][r] = bv1;
      }
  }

  // Cooperative copy of one K-stage (16 KB A + 16 KB B), fully contiguous
  // 2 KB pieces per fragment row thanks to the fragment-major global layout.
  auto issue_copy = [&](int kb, uint8_t* bA, uint8_t* bB) {
    const int k0t = kb * BKT;
#pragma unroll
    for (int it = 0; it < 4; ++it) {
      int l   = it * 4096 + tid * 16;
      int mf  = l >> 11;       // / 2048
      int rem = l & 2047;
      const uint8_t* src =
          gA + (size_t)((m_block * 8 + mf) * KT_TOT + k0t) * FRAGB + rem;
      copy16(src, bA + l);
    }
#pragma unroll
    for (int it = 0; it < 4; ++it) {
      int l   = it * 4096 + tid * 16;
      int nf  = l >> 11;
      int rem = l & 2047;
      const uint8_t* src =
          gB + (size_t)((n_block * 8 + nf) * KT_TOT + k0t) * FRAGB + rem;
      copy16(src, bB + l);
    }
  };

#if USE_ASYNC
  // Double buffer selected by pointer arithmetic only (no LDS-pointer arrays:
  // those become addrspacecast static initializers the linker rejects).
  issue_copy(0, lds, lds + STAGEB);  // 8 async b128 per wave
  for (int kb = 0; kb < NSTAGES; ++kb) {
    if (kb + 1 < NSTAGES) {
      uint8_t* nb = lds + ((kb + 1) & 1) * (2 * STAGEB);
      issue_copy(kb + 1, nb, nb + STAGEB);
      __builtin_amdgcn_s_wait_asynccnt(8);  // stage kb complete (in-order)
    } else {
      __builtin_amdgcn_s_wait_asynccnt(0);
    }
    __syncthreads();  // all waves' stage-kb LDS writes visible
    uint8_t* cb = lds + (kb & 1) * (2 * STAGEB);
    compute_stage(cb, cb + STAGEB, wm, wn, lane, c);
    __syncthreads();  // all reads done before this buffer is refilled
  }
#else
  for (int kb = 0; kb < NSTAGES; ++kb) {
    __syncthreads();
    issue_copy(kb, lds, lds + STAGEB);
    __syncthreads();
    compute_stage(lds, lds + STAGEB, wm, wn, lane, c);
  }
#endif

  // Store: C/D 16x16 f32 layout -> row = r + 8*(lane>=16), col = lane%16.
  const int row_off = (lane >> 4) * 8;
  const int col     = lane & 15;
#pragma unroll
  for (int i = 0; i < 4; ++i)
#pragma unroll
    for (int j = 0; j < 2; ++j)
#pragma unroll
      for (int r = 0; r < 8; ++r) {
        int m = m_block * BM + (wm * 4 + i) * 16 + row_off + r;
        int o = n_block * BN + (wn * 2 + j) * 16 + col;
        outp[(size_t)m * N_DIM + o] = c[i][j][r];
      }
}

// ---------------------------------------------------------------------------
extern "C" void kernel_launch(void* const* d_in, const int* in_sizes, int n_in,
                              void* d_out, int out_size, void* d_ws,
                              size_t ws_size, hipStream_t stream) {
  const float*     x     = (const float*)d_in[0];
  const int*       T     = (const int*)d_in[1];
  const float*     alpha = (const float*)d_in[2];
  const float*     mu    = (const float*)d_in[3];
  const long long* perm  = (const long long*)d_in[4];
  const float*     bias  = (const float*)d_in[5];
  float*           outp  = (float*)d_out;

  __bf16* XgF = (__bf16*)d_ws;                                          // 8 MB
  __bf16* WbF = (__bf16*)((uint8_t*)d_ws + (size_t)M_DIM * K_DIM * 2);  // +32 MB

  prep_x_kernel<<<dim3((64 * KT_TOT * 32) / 256), 256, 0, stream>>>(x, perm,
                                                                    XgF);
  prep_w_kernel<<<dim3((256 * KT_TOT * 32) / 256), 256, 0, stream>>>(T, alpha,
                                                                     mu, WbF);
  gemm_kernel<<<dim3(N_DIM / BN, M_DIM / BM), 256, 0, stream>>>(XgF, WbF, bias,
                                                                outp);
}
